// HeteroGNN_2121713844476
// MI455X (gfx1250) — compile-verified, matrix operand order checked by reading, not run on hardware
//
#include <hip/hip_runtime.h>
#include <hip/hip_bf16.h>

typedef __attribute__((ext_vector_type(16))) _Float16 v16h;
typedef __attribute__((ext_vector_type(8)))  _Float16 v8h;
typedef __attribute__((ext_vector_type(8)))  float    v8f;
typedef __attribute__((ext_vector_type(4)))  float    f4;

// explicit global-address-space pointers: force global_load/global_store (no FLAT)
typedef const float __attribute__((address_space(1)))* gcfp;
typedef       float __attribute__((address_space(1)))* gfp;
typedef const f4    __attribute__((address_space(1)))* gcf4p;
typedef       f4    __attribute__((address_space(1)))* gf4p;
typedef const int   __attribute__((address_space(1)))* gcip;

#define HDIM 128
#define NP 100000
#define NF 30000
#define EPP 1600000
#define EFP 800000
#define EPF 800000
#define NLAYERS 6
#define NGRAPH 32
#define LN_EPS 1e-5f
#define PADK 136          // padded LDS row stride in halves (272 B -> conflict-free)
#define LDS_PAIR_BYTES (128 * PADK * 2)   // 34816 B per staged weight matrix

__device__ __forceinline__ void gatomic_add(float* p, float v) {
    __hip_atomic_fetch_add((gfp)p, v, __ATOMIC_RELAXED, __HIP_MEMORY_SCOPE_AGENT);
}

// ---------------- gfx1250 async global->LDS copy (ASYNCcnt) ----------------
__device__ __forceinline__ uint32_t lds_offset_of(const void* p) {
    return (uint32_t)(size_t)(const __attribute__((address_space(3))) void*)p;
}
__device__ __forceinline__ void async_copy_b128(uint32_t lds_byte_off, const void* gsrc) {
    uint64_t ga = (uint64_t)(size_t)gsrc;
    asm volatile("global_load_async_to_lds_b128 %0, %1, off"
                 :: "v"(lds_byte_off), "v"(ga) : "memory");
}
__device__ __forceinline__ void wait_async0() {
    asm volatile("s_wait_asynccnt 0x0" ::: "memory");
}

// ---------------- WMMA tile loaders ----------------
// A operand 16x32 f16 (ISA 7.12.2): lanes 0-15 -> M rows, K runs [0..7],[16..23];
// lanes 16-31 -> same rows, K runs [8..15],[24..31]. Row scale pre-hoisted.
__device__ __forceinline__ v16h load_a_f32(const float* A, int row, int kk, float scale) {
    int lane = threadIdx.x & 31;
    int klo = kk + ((lane >> 4) << 3);
    gcfp p = (gcfp)(A + (size_t)row * HDIM + klo);
    f4 f0 = *(gcf4p)(p + 0);
    f4 f1 = *(gcf4p)(p + 4);
    f4 f2 = *(gcf4p)(p + 16);
    f4 f3 = *(gcf4p)(p + 20);
    f0 *= scale; f1 *= scale; f2 *= scale; f3 *= scale;
    v16h a;
    a[0]=(_Float16)f0[0]; a[1]=(_Float16)f0[1]; a[2]=(_Float16)f0[2]; a[3]=(_Float16)f0[3];
    a[4]=(_Float16)f1[0]; a[5]=(_Float16)f1[1]; a[6]=(_Float16)f1[2]; a[7]=(_Float16)f1[3];
    a[8]=(_Float16)f2[0]; a[9]=(_Float16)f2[1]; a[10]=(_Float16)f2[2]; a[11]=(_Float16)f2[3];
    a[12]=(_Float16)f3[0]; a[13]=(_Float16)f3[1]; a[14]=(_Float16)f3[2]; a[15]=(_Float16)f3[3];
    return a;
}

// B operand 32x16 from LDS-staged transposed weights [N][PADK]
__device__ __forceinline__ v16h load_b_lds(const _Float16* smem, int pair, int n0, int kk) {
    int lane = threadIdx.x & 31;
    int n   = n0 + (lane & 15);
    int klo = kk + ((lane >> 4) << 3);
    const _Float16* p = smem + (size_t)(pair * 128 + n) * PADK + klo;
    v8h b0 = *(const v8h*)(p);
    v8h b1 = *(const v8h*)(p + 16);
    return __builtin_shufflevector(b0, b1, 0,1,2,3,4,5,6,7,8,9,10,11,12,13,14,15);
}

// out[M x 128] = sum_p (A_p * rowscale_p) @ W_p + bias1 (+ bias2)
// Block = 8 waves; weights async-staged to LDS once per block; each wave owns a
// 16-row strip, 8 f32 accum tiles, NPAIRS*4 K-steps of v_wmma_f32_16x16x32_f16.
template <int NPAIRS>
__global__ __launch_bounds__(256) void gemm_wmma_kernel(
    const float* A0, const float* A1, const float* A2, const float* A3,
    const _Float16* W0, const _Float16* W1, const _Float16* W2, const _Float16* W3,
    const float* C0, const float* C1, const float* C2, const float* C3,
    const float* bias1, const float* bias2, float* out, int M)
{
    extern __shared__ __align__(16) _Float16 smem[];
    const float*    As[4] = {A0, A1, A2, A3};
    const _Float16* Ws[4] = {W0, W1, W2, W3};
    const float*    Cs[4] = {C0, C1, C2, C3};

    // cooperative async fill: NPAIRS matrices of 128 rows x 256B, padded to 272B rows
    const uint32_t lds_base = lds_offset_of(smem);
    for (int c = threadIdx.x; c < NPAIRS * 2048; c += 256) {
        int pair = c >> 11;
        int rem  = c & 2047;
        int n    = rem >> 4;          // row (output column) 0..127
        int ch   = rem & 15;          // 16B chunk within 256B row
        const void* gsrc = Ws[pair] + (size_t)n * HDIM + ch * 8;
        uint32_t dst = lds_base + (uint32_t)((pair * 128 + n) * PADK + ch * 8) * 2u;
        async_copy_b128(dst, gsrc);
    }
    wait_async0();
    __syncthreads();

    int wid = blockIdx.x * 8 + (threadIdx.x >> 5);
    int m0  = wid * 16;
    if (m0 >= M) return;              // whole-wave exit: EXEC stays all-ones

    // per-pair, per-row mean scale (loop-invariant): one load + one v_rcp each
    int lane = threadIdx.x & 31;
    int arow = m0 + (lane & 15);
    if (arow >= M) arow = M - 1;      // clamped rows are never stored
    float rowscale[NPAIRS];
#pragma unroll
    for (int p = 0; p < NPAIRS; ++p)
        rowscale[p] = Cs[p] ? __builtin_amdgcn_rcpf(fmaxf(*(gcfp)(Cs[p] + arow), 1.0f))
                            : 1.0f;

    v8f acc[8] = {};
#pragma unroll
    for (int ks = 0; ks < 4; ++ks) {
        int kk = ks * 32;
#pragma unroll
        for (int p = 0; p < NPAIRS; ++p) {
            v16h a = load_a_f32(As[p], arow, kk, rowscale[p]);
            v16h b[8];
#pragma unroll
            for (int nt = 0; nt < 8; ++nt) b[nt] = load_b_lds(smem, p, nt * 16, kk);
#pragma unroll
            for (int nt = 0; nt < 8; ++nt)
                acc[nt] = __builtin_amdgcn_wmma_f32_16x16x32_f16(
                    false, a, false, b[nt], (short)0, acc[nt], false, false);
        }
    }

    // C/D layout: element r of lane l -> M = m0 + r + (l>=16 ? 8:0), N = nt*16 + (l&15)
    int col  = lane & 15;
    int rofs = (lane >> 4) << 3;
#pragma unroll
    for (int nt = 0; nt < 8; ++nt) {
        int n = nt * 16 + col;
        float bv = 0.f;
        if (bias1) bv += *(gcfp)(bias1 + n);
        if (bias2) bv += *(gcfp)(bias2 + n);
#pragma unroll
        for (int r = 0; r < 8; ++r) {
            int m = m0 + r + rofs;
            if (m < M) *(gfp)(out + (size_t)m * HDIM + n) = acc[nt][r] + bv;
        }
    }
}

// ---------------- weight convert: f32 [K=128][N=128] -> f16 transposed [N][K]
__global__ void convw_kernel(const float* __restrict__ src, _Float16* __restrict__ dst) {
    int idx = blockIdx.x * blockDim.x + threadIdx.x;   // 16384 threads total
    int k = idx >> 7, n = idx & 127;
    dst[n * HDIM + k] = (_Float16)(*(gcfp)(src + idx));
}

// ---------------- misc elementwise ----------------
__global__ void zero_kernel(float* p, size_t n) {
    size_t i = blockIdx.x * (size_t)blockDim.x + threadIdx.x;
    size_t s = gridDim.x * (size_t)blockDim.x;
    for (; i < n; i += s) *(gfp)(p + i) = 0.f;
}

__global__ void enc1_kernel(const float* __restrict__ x, const float* __restrict__ W1,
                            const float* __restrict__ b1, float* __restrict__ out,
                            int M, int ID) {
    int idx = blockIdx.x * blockDim.x + threadIdx.x;
    if (idx >= M * HDIM) return;
    int m = idx >> 7, h = idx & 127;
    float a = *(gcfp)(b1 + h);
    for (int k = 0; k < ID; ++k) a += *(gcfp)(x + m * ID + k) * *(gcfp)(W1 + k * HDIM + h);
    *(gfp)(out + idx) = fmaxf(a, 0.f);
}

// wave-per-edge mean-agg scatter: lane handles 4 features; NT gather to avoid
// thrashing WGP$ with 51MB of random rows (this phase is the HBM roofline).
__global__ void scatter_kernel(const float* __restrict__ xsrc, const int* __restrict__ src,
                               const int* __restrict__ dst, float* agg, float* cnt, int E) {
    int wid  = (blockIdx.x * blockDim.x + threadIdx.x) >> 5;
    int lane = threadIdx.x & 31;
    if (wid >= E) return;
    int s = *(gcip)(src + wid);
    int d = *(gcip)(dst + wid);
    f4 v = __builtin_nontemporal_load((gcf4p)(xsrc + (size_t)s * HDIM + lane * 4));
    float* o = agg + (size_t)d * HDIM + lane * 4;
    gatomic_add(o + 0, v[0]); gatomic_add(o + 1, v[1]);
    gatomic_add(o + 2, v[2]); gatomic_add(o + 3, v[3]);
    if (lane == 0) gatomic_add(cnt + d, 1.0f);
}

// h = LayerNorm(h + relu(o)) * gamma + beta, wave-per-row (wave32 shuffles)
__global__ void resid_relu_ln_kernel(float* h, const float* __restrict__ o,
                                     const float* __restrict__ gamma,
                                     const float* __restrict__ beta, int M) {
    int wid  = (blockIdx.x * blockDim.x + threadIdx.x) >> 5;
    int lane = threadIdx.x & 31;
    if (wid >= M) return;
    float* hp = h + (size_t)wid * HDIM + lane * 4;
    f4 hv = *(gcf4p)hp;
    f4 ov = *(gcf4p)(o + (size_t)wid * HDIM + lane * 4);
    float y0 = hv[0] + fmaxf(ov[0], 0.f), y1 = hv[1] + fmaxf(ov[1], 0.f);
    float y2 = hv[2] + fmaxf(ov[2], 0.f), y3 = hv[3] + fmaxf(ov[3], 0.f);
    float s = y0 + y1 + y2 + y3;
    for (int off = 16; off >= 1; off >>= 1) s += __shfl_xor(s, off, 32);
    float mean = s * (1.0f / 128.0f);
    float d0 = y0 - mean, d1 = y1 - mean, d2 = y2 - mean, d3 = y3 - mean;
    float v = d0 * d0 + d1 * d1 + d2 * d2 + d3 * d3;
    for (int off = 16; off >= 1; off >>= 1) v += __shfl_xor(v, off, 32);
    float rstd = rsqrtf(v * (1.0f / 128.0f) + LN_EPS);
    f4 gv = *(gcf4p)(gamma + lane * 4);
    f4 bv = *(gcf4p)(beta + lane * 4);
    f4 r;
    r[0] = d0 * rstd * gv[0] + bv[0]; r[1] = d1 * rstd * gv[1] + bv[1];
    r[2] = d2 * rstd * gv[2] + bv[2]; r[3] = d3 * rstd * gv[3] + bv[3];
    *(gf4p)hp = r;
}

// wave-per-row pooling scatter into g[B x 256] (+ per-type count)
__global__ void pool_scatter_kernel(const float* __restrict__ h, const int* __restrict__ batch,
                                    float* g, float* cnt, int M, int coloff) {
    int wid  = (blockIdx.x * blockDim.x + threadIdx.x) >> 5;
    int lane = threadIdx.x & 31;
    if (wid >= M) return;
    int b = *(gcip)(batch + wid);
    f4 v = *(gcf4p)(h + (size_t)wid * HDIM + lane * 4);
    float* o = g + b * 256 + coloff + lane * 4;
    gatomic_add(o + 0, v[0]); gatomic_add(o + 1, v[1]);
    gatomic_add(o + 2, v[2]); gatomic_add(o + 3, v[3]);
    if (lane == 0) gatomic_add(cnt + b, 1.0f);
}

// tiny decoder: [32 x 256] @ [256x128] relu @ [128x128] relu @ [128x1]
__global__ __launch_bounds__(128) void decoder_kernel(
    const float* __restrict__ g, const float* __restrict__ cnt,
    const float* __restrict__ dW1, const float* __restrict__ db1,
    const float* __restrict__ dW2, const float* __restrict__ db2,
    const float* __restrict__ dW3, const float* __restrict__ db3, float* out) {
    __shared__ float gb[NGRAPH * 256];
    __shared__ float h1[128];
    __shared__ float h2[128];
    int t = threadIdx.x;
    for (int idx = t; idx < NGRAPH * 256; idx += 128) {
        int bi = idx >> 8, c = idx & 255;
        float cc = (c < 128) ? cnt[bi] : cnt[NGRAPH + bi];
        gb[idx] = g[idx] / fmaxf(cc, 1.0f);
    }
    __syncthreads();
    for (int bi = 0; bi < NGRAPH; ++bi) {
        float a = db1[t];
        for (int k = 0; k < 256; ++k) a += gb[bi * 256 + k] * dW1[k * 128 + t];
        h1[t] = fmaxf(a, 0.f);
        __syncthreads();
        float a2 = db2[t];
        for (int k = 0; k < 128; ++k) a2 += h1[k] * dW2[k * 128 + t];
        h2[t] = fmaxf(a2, 0.f);
        __syncthreads();
        if (t == 0) {
            float s = db3[0];
            for (int k = 0; k < 128; ++k) s += h2[k] * dW3[k];
            out[bi] = s;
        }
        __syncthreads();
    }
}

// ---------------- host orchestration ----------------
extern "C" void kernel_launch(void* const* d_in, const int* in_sizes, int n_in,
                              void* d_out, int out_size, void* d_ws, size_t ws_size,
                              hipStream_t stream) {
    const float* x_point = (const float*)d_in[0];
    const float* x_face  = (const float*)d_in[1];
    const int* epp_src = (const int*)d_in[2];
    const int* epp_dst = (const int*)d_in[3];
    const int* efp_src = (const int*)d_in[4];
    const int* efp_dst = (const int*)d_in[5];
    const int* epf_src = (const int*)d_in[6];
    const int* epf_dst = (const int*)d_in[7];
    const int* batch_point = (const int*)d_in[8];
    const int* batch_face  = (const int*)d_in[9];
    const float* pe_W1 = (const float*)d_in[11]; const float* pe_b1 = (const float*)d_in[12];
    const float* pe_W2 = (const float*)d_in[13]; const float* pe_b2 = (const float*)d_in[14];
    const float* fe_W1 = (const float*)d_in[15]; const float* fe_b1 = (const float*)d_in[16];
    const float* fe_W2 = (const float*)d_in[17]; const float* fe_b2 = (const float*)d_in[18];
    const float* Wl_pp = (const float*)d_in[19]; const float* bl_pp = (const float*)d_in[20];
    const float* Wr_pp = (const float*)d_in[21];
    const float* Wl_fp = (const float*)d_in[22]; const float* bl_fp = (const float*)d_in[23];
    const float* Wr_fp = (const float*)d_in[24];
    const float* Wl_pf = (const float*)d_in[25]; const float* bl_pf = (const float*)d_in[26];
    const float* Wr_pf = (const float*)d_in[27];
    const float* ln_p_g = (const float*)d_in[28]; const float* ln_p_b = (const float*)d_in[29];
    const float* ln_f_g = (const float*)d_in[30]; const float* ln_f_b = (const float*)d_in[31];
    const float* dW1 = (const float*)d_in[32]; const float* db1 = (const float*)d_in[33];
    const float* dW2 = (const float*)d_in[34]; const float* db2 = (const float*)d_in[35];
    const float* dW3 = (const float*)d_in[36]; const float* db3 = (const float*)d_in[37];
    (void)in_sizes; (void)n_in; (void)out_size; (void)ws_size;

    // allow >64KB dynamic LDS for the 2- and 4-matrix instantiations
    static bool attr_done = [] {
        hipFuncSetAttribute(reinterpret_cast<const void*>(gemm_wmma_kernel<1>),
                            hipFuncAttributeMaxDynamicSharedMemorySize, 1 * LDS_PAIR_BYTES);
        hipFuncSetAttribute(reinterpret_cast<const void*>(gemm_wmma_kernel<2>),
                            hipFuncAttributeMaxDynamicSharedMemorySize, 2 * LDS_PAIR_BYTES);
        hipFuncSetAttribute(reinterpret_cast<const void*>(gemm_wmma_kernel<4>),
                            hipFuncAttributeMaxDynamicSharedMemorySize, 4 * LDS_PAIR_BYTES);
        return true;
    }();
    (void)attr_done;

    // workspace layout (f32)
    float* ws = (float*)d_ws;
    float* hp     = ws;                      // NP*128
    float* hf     = hp + (size_t)NP * HDIM;  // NF*128
    float* outp   = hf + (size_t)NF * HDIM;  // NP*128 (also encoder tmp for points)
    float* outf   = outp + (size_t)NP * HDIM;// NF*128 (also encoder tmp for faces)
    float* agg_pp = outf + (size_t)NF * HDIM;// NP*128
    float* agg_fp = agg_pp + (size_t)NP * HDIM; // NP*128
    float* agg_pf = agg_fp + (size_t)NP * HDIM; // NF*128
    float* cnt_pp = agg_pf + (size_t)NF * HDIM; // NP
    float* cnt_fp = cnt_pp + NP;                // NP
    float* cnt_pf = cnt_fp + NP;                // NF
    float* pool_g   = cnt_pf + NF;              // 32*256
    float* pool_cnt = pool_g + NGRAPH * 256;    // 64 (point counts, then face)
    _Float16* wf16 = (_Float16*)(pool_cnt + 64);
    const size_t WSLOT = (size_t)HDIM * HDIM;   // 16384 halves per matrix
    auto wt = [&](int slot) { return wf16 + (size_t)slot * WSLOT; };

    // 1) convert+transpose all 128x128 weights to f16 [N][K]
    auto convw = [&](const float* src, int slot) {
        convw_kernel<<<64, 256, 0, stream>>>(src, wt(slot));
    };
    convw(pe_W2, 0);
    convw(fe_W2, 1);
    for (int i = 0; i < NLAYERS; ++i) {
        int base = 2 + i * 6;
        convw(Wl_pp + (size_t)i * WSLOT, base + 0);
        convw(Wr_pp + (size_t)i * WSLOT, base + 1);
        convw(Wl_fp + (size_t)i * WSLOT, base + 2);
        convw(Wr_fp + (size_t)i * WSLOT, base + 3);
        convw(Wl_pf + (size_t)i * WSLOT, base + 4);
        convw(Wr_pf + (size_t)i * WSLOT, base + 5);
    }

    auto grid_for = [](int M) { return ((M + 15) / 16 + 7) / 8; };

    // 2) encoders
    enc1_kernel<<<(NP * HDIM + 255) / 256, 256, 0, stream>>>(x_point, pe_W1, pe_b1, outp, NP, 3);
    gemm_wmma_kernel<1><<<grid_for(NP), 256, 1 * LDS_PAIR_BYTES, stream>>>(
        outp, nullptr, nullptr, nullptr, wt(0), nullptr, nullptr, nullptr,
        nullptr, nullptr, nullptr, nullptr, pe_b2, nullptr, hp, NP);
    enc1_kernel<<<(NF * HDIM + 255) / 256, 256, 0, stream>>>(x_face, fe_W1, fe_b1, outf, NF, 8);
    gemm_wmma_kernel<1><<<grid_for(NF), 256, 1 * LDS_PAIR_BYTES, stream>>>(
        outf, nullptr, nullptr, nullptr, wt(1), nullptr, nullptr, nullptr,
        nullptr, nullptr, nullptr, nullptr, fe_b2, nullptr, hf, NF);

    // 3) message-passing layers (mean-normalization fused into GEMM A loads)
    const size_t agg_zero_count =
        (size_t)NP * HDIM * 2 + (size_t)NF * HDIM + (size_t)NP * 2 + NF;
    for (int i = 0; i < NLAYERS; ++i) {
        int base = 2 + i * 6;
        zero_kernel<<<2048, 256, 0, stream>>>(agg_pp, agg_zero_count);
        scatter_kernel<<<(EPP + 7) / 8, 256, 0, stream>>>(hp, epp_src, epp_dst, agg_pp, cnt_pp, EPP);
        scatter_kernel<<<(EFP + 7) / 8, 256, 0, stream>>>(hf, efp_src, efp_dst, agg_fp, cnt_fp, EFP);
        scatter_kernel<<<(EPF + 7) / 8, 256, 0, stream>>>(hp, epf_src, epf_dst, agg_pf, cnt_pf, EPF);
        // out_p = (agg_pp/deg)@Wl_pp + hp@Wr_pp + (agg_fp/deg)@Wl_fp + hp@Wr_fp + bl_pp + bl_fp
        gemm_wmma_kernel<4><<<grid_for(NP), 256, 4 * LDS_PAIR_BYTES, stream>>>(
            agg_pp, hp, agg_fp, hp,
            wt(base + 0), wt(base + 1), wt(base + 2), wt(base + 3),
            cnt_pp, nullptr, cnt_fp, nullptr,
            bl_pp + (size_t)i * HDIM, bl_fp + (size_t)i * HDIM, outp, NP);
        // out_f = (agg_pf/deg)@Wl_pf + hf@Wr_pf + bl_pf
        gemm_wmma_kernel<2><<<grid_for(NF), 256, 2 * LDS_PAIR_BYTES, stream>>>(
            agg_pf, hf, nullptr, nullptr,
            wt(base + 4), wt(base + 5), nullptr, nullptr,
            cnt_pf, nullptr, nullptr, nullptr,
            bl_pf + (size_t)i * HDIM, nullptr, outf, NF);
        resid_relu_ln_kernel<<<(NP * 32 + 255) / 256, 256, 0, stream>>>(
            hp, outp, ln_p_g + (size_t)i * HDIM, ln_p_b + (size_t)i * HDIM, NP);
        resid_relu_ln_kernel<<<(NF * 32 + 255) / 256, 256, 0, stream>>>(
            hf, outf, ln_f_g + (size_t)i * HDIM, ln_f_b + (size_t)i * HDIM, NF);
    }

    // 4) global mean pool + decoder
    zero_kernel<<<64, 256, 0, stream>>>(pool_g, (size_t)NGRAPH * 256 + 64);
    pool_scatter_kernel<<<(NP * 32 + 255) / 256, 256, 0, stream>>>(
        hp, batch_point, pool_g, pool_cnt, NP, 0);
    pool_scatter_kernel<<<(NF * 32 + 255) / 256, 256, 0, stream>>>(
        hf, batch_face, pool_g, pool_cnt + NGRAPH, NF, 128);
    decoder_kernel<<<1, 128, 0, stream>>>(pool_g, pool_cnt, dW1, db1, dW2, db2, dW3, db3,
                                          (float*)d_out);
}
